// NeptuneMoEModel_68831145886459
// MI455X (gfx1250) — matrix operand order repaired and checked by reading, not compiled, not run on hardware
//
#include <hip/hip_runtime.h>
#include <hip/hip_bf16.h>

typedef _Float16 v16h __attribute__((ext_vector_type(16)));
typedef _Float16 v8h  __attribute__((ext_vector_type(8)));
typedef float    v8f  __attribute__((ext_vector_type(8)));

#define BB   2048      // batches
#define TT   256       // tokens per batch
#define DD   256       // hidden dim
#define HH   512       // head hidden
#define KPAD 288       // 259 padded to multiple of 32
#define NHID 3072      // 6 heads * 512
#define SROW 264       // LDS row stride in halves (256 + 8 pad -> bank rotation)
#define TSUB 64        // rows per main-GEMM workgroup

struct Ptr6    { const float* p[6]; };
struct HeadOut { const float* W2[6]; const float* b2[6]; };

// ---- branch-free transcendentals (v_exp_f32 / v_rcp_f32, no libm exec-mask games) ----
#define LOG2E 1.4426950408889634f

__device__ __forceinline__ float fast_exp(float x) {
  return __builtin_amdgcn_exp2f(x * LOG2E);
}
__device__ __forceinline__ float fast_tanh(float x) {
  float ax = fabsf(x);
  float e  = __builtin_amdgcn_exp2f(ax * (-2.0f * LOG2E));   // exp(-2|x|)
  float r  = (1.0f - e) * __builtin_amdgcn_rcpf(1.0f + e);
  return copysignf(r, x);
}
__device__ __forceinline__ float gelu_tanh(float x) {
  float inner = 0.7978845608028654f * (x + 0.044715f * x * x * x);
  return 0.5f * x * (1.0f + fast_tanh(inner));
}
__device__ __forceinline__ float fast_sigmoid(float x) {
  return __builtin_amdgcn_rcpf(1.0f + __builtin_amdgcn_exp2f(-x * LOG2E));
}

// A-fragment (16x32 f16) from LDS tile, row-major, stride SROW halves.
// lane l<16: row=l,    K = {k0..k0+7, k0+16..k0+23}, k0 = kt*32
// lane l>=16: row=l-16, same but k0 = kt*32+8
__device__ __forceinline__ v16h frag_a_lds(const _Float16* base, int mBase, int kt, int lane) {
  int row = mBase + (lane & 15);
  int k0  = kt * 32 + ((lane >> 4) << 3);
  union { v16h v; v8h h[2]; } u;
  u.h[0] = *(const v8h*)(base + row * SROW + k0);
  u.h[1] = *(const v8h*)(base + row * SROW + k0 + 16);
  return u.v;
}

// A-fragment from global f16 row-major [M][ld]
__device__ __forceinline__ v16h frag_a_glb(const _Float16* A, int mBase, int kt, int lane, int ld) {
  int row = mBase + (lane & 15);
  int k0  = kt * 32 + ((lane >> 4) << 3);
  const _Float16* p = A + (size_t)row * ld + k0;
  union { v16h v; v8h h[2]; } u;
  u.h[0] = *(const v8h*)p;
  u.h[1] = *(const v8h*)(p + 16);
  return u.v;
}

// B-fragment (32x16 f16) from pre-TRANSPOSED weights Bt[n][k], stride ld halves.
// lane l<16: col=l, K = {k0..k0+7, k0+16..k0+23}, k0 = kt*32 ; lane>=16: k0 = kt*32+8
__device__ __forceinline__ v16h frag_b_glb(const _Float16* Bt, int nBase, int kt, int lane, int ld) {
  int col = nBase + (lane & 15);
  int k0  = kt * 32 + ((lane >> 4) << 3);
  const _Float16* p = Bt + (size_t)col * ld + k0;
  union { v16h v; v8h h[2]; } u;
  u.h[0] = *(const v8h*)p;
  u.h[1] = *(const v8h*)(p + 16);
  return u.v;
}

// ---------------- prep kernels ----------------

__global__ void k_zero(float* p, long n) {
  long i = (long)blockIdx.x * blockDim.x + threadIdx.x;
  if (i < n) p[i] = 0.0f;
}

// W_h [k][n] f32  ->  Wh_t [n][k] f16
__global__ void k_prep_wh(const float* Wh, _Float16* Wh_t) {
  int n = blockIdx.x, k = threadIdx.x;
  Wh_t[(size_t)n * DD + k] = (_Float16)Wh[(size_t)k * DD + n];
}

// six W1 [259][512] f32  ->  W1t [3072][288] f16 (transposed, zero-padded K)
__global__ void k_prep_w1(Ptr6 w1, _Float16* W1t) {
  int row = blockIdx.x;      // 0..3071 = head*512 + n
  int k   = threadIdx.x;     // 0..287
  int hd = row >> 9, n = row & 511;
  float v = (k < 259) ? w1.p[hd][(size_t)k * HH + n] : 0.0f;
  W1t[(size_t)row * KPAD + k] = (_Float16)v;
}

// ---------------- fused layer1 + layer2 WMMA + pooling ----------------
// One workgroup = 64 rows of one batch. 256 threads = 8 waves.
__global__ void __launch_bounds__(256)
k_main(const float* __restrict__ coords, const float* __restrict__ feats,
       const float* __restrict__ W_in, const float* __restrict__ b_in,
       const _Float16* __restrict__ Wh_t, const float* __restrict__ b_h,
       float* __restrict__ pooled_sum, float* __restrict__ cent_sum) {
  __shared__ _Float16 h1s[TSUB * SROW];   // 33792 B
  __shared__ float    pts[TSUB * 12];     //  3072 B
  __shared__ float    colsum[DD];         //  1024 B
  __shared__ float    csum[3];

  int tid = threadIdx.x;
  int wg  = blockIdx.x;         // 0 .. BB*4-1
  int b   = wg >> 2;
  int sub = wg & 3;
  long rowBase = (long)b * TT + sub * TSUB;

  colsum[tid & 255] = 0.0f;
  if (tid < 3) csum[tid] = 0.0f;

  // stage pts rows (3 coords + 6 features) into LDS
  if (tid < TSUB) {
    long r = rowBase + tid;
    float* p = &pts[tid * 12];
    p[0] = coords[r * 3 + 0]; p[1] = coords[r * 3 + 1]; p[2] = coords[r * 3 + 2];
    #pragma unroll
    for (int j = 0; j < 6; ++j) p[3 + j] = feats[r * 6 + j];
  }
  __syncthreads();

  // centroid partial sums (64 rows * 3 comps = 192 adds)
  if (tid < 192) atomicAdd(&csum[tid % 3], pts[(tid / 3) * 12 + (tid % 3)]);

  // layer 1: thread owns column tid for all 64 rows, gelu, f16 into LDS
  {
    float w[9];
    #pragma unroll
    for (int j = 0; j < 9; ++j) w[j] = W_in[j * DD + tid];
    float bb = b_in[tid];
    for (int i = 0; i < TSUB; ++i) {
      const float* p = &pts[i * 12];
      float acc = bb;
      #pragma unroll
      for (int j = 0; j < 9; ++j) acc += p[j] * w[j];
      h1s[i * SROW + tid] = (_Float16)gelu_tanh(acc);
    }
  }
  __syncthreads();

  // layer 2 WMMA: wave w -> mtile (w&3), ntiles ((w>>2)*8 .. +7)
  int wave = tid >> 5, lane = tid & 31;
  int mt   = wave & 3;
  int ng0  = (wave >> 2) * 8;
  for (int nt = ng0; nt < ng0 + 8; ++nt) {
    v8f acc = {};
    #pragma unroll
    for (int kt = 0; kt < 8; ++kt) {
      v16h a  = frag_a_lds(h1s, mt * 16, kt, lane);
      v16h bf = frag_b_glb(Wh_t, nt * 16, kt, lane, DD);
      acc = __builtin_amdgcn_wmma_f32_16x16x32_f16(false, a, false, bf,
                                                   (short)0, acc, false, false);
    }
    int col = nt * 16 + (lane & 15);
    float bias = b_h[col];
    float cs = 0.0f;
    #pragma unroll
    for (int r = 0; r < 8; ++r) cs += gelu_tanh(acc[r] + bias);
    atomicAdd(&colsum[col], cs);   // h2 never leaves the WGP
  }
  __syncthreads();

  atomicAdd(&pooled_sum[(size_t)b * DD + tid], colsum[tid]);
  if (tid < 3) atomicAdd(&cent_sum[(size_t)b * 3 + tid], csum[tid]);
}

// build x_pad f16 [BB][288] = [pooled(256) | centroid(3) | zeros(29)]
__global__ void k_xpad(const float* pooled_sum, const float* cent_sum, _Float16* x_pad) {
  int b = blockIdx.x, t = threadIdx.x;   // 288 threads
  float v;
  if (t < DD)       v = pooled_sum[(size_t)b * DD + t] * (1.0f / 256.0f);
  else if (t < 259) v = cent_sum[(size_t)b * 3 + (t - 256)] * (1.0f / 256.0f);
  else              v = 0.0f;
  x_pad[(size_t)b * KPAD + t] = (_Float16)v;
}

// heads hidden GEMM: [2048 x 288] @ [288 x 3072] -> gelu -> f16 hidden
// grid (48, 16), 256 thr; wave owns 16 rows x 64 cols, K-loop 9 tiles.
__global__ void __launch_bounds__(256)
k_heads(const _Float16* __restrict__ x_pad, const _Float16* __restrict__ W1t,
        Ptr6 b1, _Float16* __restrict__ hidden) {
  int wave = threadIdx.x >> 5, lane = threadIdx.x & 31;
  int mt = blockIdx.y * 8 + wave;    // 0..127
  int nb = blockIdx.x * 4;           // ntile base
  v8f acc[4];
  v8f z = {};
  #pragma unroll
  for (int i = 0; i < 4; ++i) acc[i] = z;

  for (int kt = 0; kt < 9; ++kt) {
    v16h a = frag_a_glb(x_pad, mt * 16, kt, lane, KPAD);
    #pragma unroll
    for (int i = 0; i < 4; ++i) {
      v16h bf = frag_b_glb(W1t, (nb + i) * 16, kt, lane, KPAD);
      acc[i] = __builtin_amdgcn_wmma_f32_16x16x32_f16(false, a, false, bf,
                                                      (short)0, acc[i], false, false);
    }
  }
  int rbase = mt * 16 + ((lane >> 4) << 3);
  #pragma unroll
  for (int i = 0; i < 4; ++i) {
    int col = (nb + i) * 16 + (lane & 15);
    float bias = b1.p[col >> 9][col & 511];
    #pragma unroll
    for (int r = 0; r < 8; ++r)
      hidden[(size_t)(rbase + r) * NHID + col] = (_Float16)gelu_tanh(acc[i][r] + bias);
  }
}

// second head layers (K=512, 19 outs) + softmax/sigmoid gating; one WG per row
__global__ void __launch_bounds__(256)
k_out(const _Float16* __restrict__ hidden, HeadOut hp, float* __restrict__ out) {
  __shared__ float red[256];
  __shared__ float sv[19];
  int row = blockIdx.x, t = threadIdx.x;
  const int odim[6] = {6, 2, 2, 3, 3, 3};
  const int ooff[6] = {0, 6, 8, 10, 13, 16};

  float part[19];
  #pragma unroll
  for (int i = 0; i < 19; ++i) part[i] = 0.0f;

  const _Float16* hrow = hidden + (size_t)row * NHID;
  #pragma unroll
  for (int h = 0; h < 6; ++h) {
    float a = (float)hrow[h * 512 + t];
    float c = (float)hrow[h * 512 + 256 + t];
    const float* W2 = hp.W2[h];
    int od = odim[h], off = ooff[h];
    for (int o = 0; o < od; ++o)
      part[off + o] += a * W2[t * od + o] + c * W2[(t + 256) * od + o];
  }
  for (int o = 0; o < 19; ++o) {
    red[t] = part[o];
    __syncthreads();
    for (int s = 128; s > 0; s >>= 1) {
      if (t < s) red[t] += red[t + s];
      __syncthreads();
    }
    if (t == 0) sv[o] = red[0];
    __syncthreads();
  }
  if (t == 0) {
    float lg[6];
    #pragma unroll
    for (int i = 0; i < 6; ++i) lg[i] = sv[i] + hp.b2[0][i];
    float ec0 = sv[6] + hp.b2[1][0], ec1 = sv[7] + hp.b2[1][1];
    float eu0 = sv[8] + hp.b2[2][0], eu1 = sv[9] + hp.b2[2][1];
    float dc[3], dl[3], dh[3];
    #pragma unroll
    for (int i = 0; i < 3; ++i) {
      dc[i] = sv[10 + i] + hp.b2[3][i];
      dl[i] = sv[13 + i] + hp.b2[4][i];
      dh[i] = sv[16 + i] + hp.b2[5][i];
    }
    float mx = lg[0];
    #pragma unroll
    for (int i = 1; i < 6; ++i) mx = fmaxf(mx, lg[i]);
    float ex[6], s = 0.0f;
    #pragma unroll
    for (int i = 0; i < 6; ++i) { ex[i] = fast_exp(lg[i] - mx); s += ex[i]; }
    float inv = __builtin_amdgcn_rcpf(s);
    float pr[6];
    #pragma unroll
    for (int i = 0; i < 6; ++i) pr[i] = fmaxf(ex[i] * inv, 1e-6f);
    float p_cont = pr[0] + pr[1];
    float p_unc  = pr[2] + pr[3] + pr[5];
    float e0 = p_cont * ec0 + p_unc * eu0;
    float e1 = p_cont * ec1 + p_unc * eu1;
    float p_cas = pr[0];
    float p_trk = pr[1] + pr[2] + pr[3] + pr[5];
    float gate = fast_sigmoid(e0 - 4.0f);  // log10(10000) = 4
    float* o = out + (size_t)row * 11;
    #pragma unroll
    for (int i = 0; i < 6; ++i) o[i] = lg[i];
    o[6] = e0; o[7] = e1;
    #pragma unroll
    for (int i = 0; i < 3; ++i)
      o[8 + i] = p_cas * dc[i] + p_trk * (1.0f - gate) * dl[i] + p_trk * gate * dh[i];
  }
}

extern "C" void kernel_launch(void* const* d_in, const int* in_sizes, int n_in,
                              void* d_out, int out_size, void* d_ws, size_t ws_size,
                              hipStream_t stream) {
  (void)in_sizes; (void)n_in; (void)out_size; (void)ws_size;
  const float* coords = (const float*)d_in[0];
  const float* feats  = (const float*)d_in[1];
  const float* W_in   = (const float*)d_in[3];
  const float* b_in   = (const float*)d_in[4];
  const float* W_h    = (const float*)d_in[5];
  const float* b_h    = (const float*)d_in[6];
  Ptr6 w1, b1; HeadOut ho;
  for (int i = 0; i < 6; ++i) {
    w1.p[i]  = (const float*)d_in[7 + 4 * i];
    b1.p[i]  = (const float*)d_in[8 + 4 * i];
    ho.W2[i] = (const float*)d_in[9 + 4 * i];
    ho.b2[i] = (const float*)d_in[10 + 4 * i];
  }

  char* ws = (char*)d_ws;
  size_t off = 0;
  auto alloc = [&](size_t bytes) -> void* {
    void* p = ws + off;
    off = (off + bytes + 255) & ~(size_t)255;
    return p;
  };
  _Float16* Wh_t   = (_Float16*)alloc((size_t)DD * DD * 2);         // 128 KB
  _Float16* W1t    = (_Float16*)alloc((size_t)NHID * KPAD * 2);     // 1.7 MB
  _Float16* x_pad  = (_Float16*)alloc((size_t)BB * KPAD * 2);       // 1.1 MB
  _Float16* hidden = (_Float16*)alloc((size_t)BB * NHID * 2);       // 12.6 MB
  float*    pooled = (float*)alloc((size_t)BB * (DD + 3) * 4);      // pooled | cent
  float*    cent   = pooled + (size_t)BB * DD;
  float*    out    = (float*)d_out;

  long nz = (long)BB * (DD + 3);
  k_zero<<<(int)((nz + 255) / 256), 256, 0, stream>>>(pooled, nz);
  k_prep_wh<<<DD, DD, 0, stream>>>(W_h, Wh_t);
  k_prep_w1<<<NHID, KPAD, 0, stream>>>(w1, W1t);
  k_main<<<BB * (TT / TSUB), 256, 0, stream>>>(coords, feats, W_in, b_in,
                                               Wh_t, b_h, pooled, cent);
  k_xpad<<<BB, KPAD, 0, stream>>>(pooled, cent, x_pad);
  k_heads<<<dim3(NHID / 64, 2048 / 128), 256, 0, stream>>>(x_pad, W1t, b1, hidden);
  k_out<<<BB, 256, 0, stream>>>(hidden, ho, out);
}
